// AdaptiveFourierPositionEncoding_80831284511311
// MI455X (gfx1250) — matrix operand order repaired and checked by reading, not compiled, not run on hardware
//
#include <hip/hip_runtime.h>

typedef __attribute__((ext_vector_type(16))) _Float16 v16h;
typedef __attribute__((ext_vector_type(8)))  float    v8f;

#define BB 8
#define SS 32768
#define DIM 128
#define NB 64
#define TOKENS (BB * SS)
#define WAVES_PER_BLOCK 8
#define TOK_PER_WAVE 16
#define TOK_PER_BLOCK (WAVES_PER_BLOCK * TOK_PER_WAVE)

__global__ __launch_bounds__(256) void afpe_kernel(
    const float* __restrict__ x,
    const int*   __restrict__ positions,
    const float* __restrict__ freq,
    const float* __restrict__ phase,
    const float* __restrict__ attn_w,
    const float* __restrict__ attn_b,
    float* __restrict__ out)
{
    // attn_w staged as f16 (WMMA B operand source), tiny vectors, per-wave logits
    __shared__ __align__(32) _Float16 wlds[NB * DIM];                 // 16 KB
    __shared__ float fb[NB], ph[NB], bb[NB];                          // 768 B
    __shared__ float logits[WAVES_PER_BLOCK][TOK_PER_WAVE][NB];       // 32 KB

    const int tid  = threadIdx.x;
    const int wave = tid >> 5;
    const int lane = tid & 31;
    const int g    = lane >> 4;   // lane group (0: lanes 0-15, 1: lanes 16-31)
    const int ln   = lane & 15;

    // ---- stage attn_w f32->f16 into LDS; freq/phase/bias too ----
    {
        const float4* w4 = (const float4*)attn_w;
        const int base = tid * 32;
#pragma unroll
        for (int j = 0; j < 32; j += 4) {
            float4 f = w4[(base + j) >> 2];
            wlds[base + j + 0] = (_Float16)f.x;
            wlds[base + j + 1] = (_Float16)f.y;
            wlds[base + j + 2] = (_Float16)f.z;
            wlds[base + j + 3] = (_Float16)f.w;
        }
        if (tid < NB) { fb[tid] = freq[tid]; ph[tid] = phase[tid]; bb[tid] = attn_b[tid]; }
    }

    const int token0 = blockIdx.x * TOK_PER_BLOCK + wave * TOK_PER_WAVE;

    // ---- load A fragments: 16 rows of x, f32 -> f16, WMMA 16x32 A layout ----
    // lane holds row M = ln; halves 0-15: K = k0 + g*16 + (0..15)
    const float* xrow = x + (size_t)(token0 + ln) * DIM;

    // gfx1250 prefetch of the next block's tile (global_prefetch_b8)
    if (blockIdx.x + 1 < gridDim.x)
        __builtin_prefetch(xrow + (size_t)TOK_PER_BLOCK * DIM, 0, 1);

    v16h a[4];
#pragma unroll
    for (int kc = 0; kc < 4; ++kc) {
        const int k0 = kc * 32;
        float4 r0 = *(const float4*)(xrow + k0 + g * 16);
        float4 r1 = *(const float4*)(xrow + k0 + g * 16 + 4);
        float4 r2 = *(const float4*)(xrow + k0 + g * 16 + 8);
        float4 r3 = *(const float4*)(xrow + k0 + g * 16 + 12);
        a[kc][0]  = (_Float16)r0.x;  a[kc][1]  = (_Float16)r0.y;
        a[kc][2]  = (_Float16)r0.z;  a[kc][3]  = (_Float16)r0.w;
        a[kc][4]  = (_Float16)r1.x;  a[kc][5]  = (_Float16)r1.y;
        a[kc][6]  = (_Float16)r1.z;  a[kc][7]  = (_Float16)r1.w;
        a[kc][8]  = (_Float16)r2.x;  a[kc][9]  = (_Float16)r2.y;
        a[kc][10] = (_Float16)r2.z;  a[kc][11] = (_Float16)r2.w;
        a[kc][12] = (_Float16)r3.x;  a[kc][13] = (_Float16)r3.y;
        a[kc][14] = (_Float16)r3.z;  a[kc][15] = (_Float16)r3.w;
    }

    __syncthreads();   // wlds / fb / ph / bb ready

    // ---- logits = x @ attn_w^T + b via V_WMMA_F32_16X16X32_F16 ----
    // B fragment layout: lane (g,n=ln), halves 0..15 hold K = k0 + g*16 + (0..15), col n
#pragma unroll
    for (int nt = 0; nt < 4; ++nt) {
        const int n0 = nt * 16;
        v8f acc = {};
#pragma unroll
        for (int kc = 0; kc < 4; ++kc) {
            const v16h bfrag = *(const v16h*)&wlds[(n0 + ln) * DIM + kc * 32 + g * 16];
            acc = __builtin_amdgcn_wmma_f32_16x16x32_f16(
                false, a[kc], false, bfrag, (short)0, acc, false, false);
        }
        // C layout: VGPR c -> row (g*8 + c), col ln
        const float bias = bb[n0 + ln];
#pragma unroll
        for (int c = 0; c < 8; ++c)
            logits[wave][g * 8 + c][n0 + ln] = acc[c] + bias;
    }

    __syncthreads();   // logits visible

    // ---- softmax over 64 bands + fourier encoding + fused output ----
    // 2 lanes per token: lane = tok*2 + p ; each lane owns 32 bands -> 64 output dims
    const int tok = lane >> 1;
    const int p   = lane & 1;
    const float* lg = &logits[wave][tok][p * 32];

    float l[32];
    float m = -3.4e38f;
#pragma unroll
    for (int i = 0; i < 32; ++i) { l[i] = lg[i]; m = fmaxf(m, l[i]); }
    // combine with partner lane: ds_swizzle SWAPX1 (xor=1, and=0x1f) = 0x041F
    m = fmaxf(m, __int_as_float(__builtin_amdgcn_ds_swizzle(__float_as_int(m), 0x041F)));

    float s = 0.f;
#pragma unroll
    for (int i = 0; i < 32; ++i) { l[i] = __expf(l[i] - m); s += l[i]; }
    s += __int_as_float(__builtin_amdgcn_ds_swizzle(__float_as_int(s), 0x041F));
    const float inv = __builtin_amdgcn_rcpf(s);

    const int   tglob = token0 + tok;
    const float pos   = (float)positions[tglob];
    const float4* xr  = (const float4*)(x   + (size_t)tglob * DIM + p * 64);
    float4*      outr = (float4*)      (out + (size_t)tglob * DIM + p * 64);

#pragma unroll
    for (int i = 0; i < 32; i += 2) {
        const int b0 = p * 32 + i;
        const float w0 = l[i]     * inv;
        const float w1 = l[i + 1] * inv;
        const float a0 = fmaf(pos, fb[b0],     ph[b0]);
        const float a1 = fmaf(pos, fb[b0 + 1], ph[b0 + 1]);
        float4 xv = xr[i >> 1];
        float4 ov;
        ov.x = fmaf(__sinf(a0), w0, xv.x);
        ov.y = fmaf(__cosf(a0), w0, xv.y);
        ov.z = fmaf(__sinf(a1), w1, xv.z);
        ov.w = fmaf(__cosf(a1), w1, xv.w);
        outr[i >> 1] = ov;
    }
}

extern "C" void kernel_launch(void* const* d_in, const int* in_sizes, int n_in,
                              void* d_out, int out_size, void* d_ws, size_t ws_size,
                              hipStream_t stream) {
    (void)in_sizes; (void)n_in; (void)out_size; (void)d_ws; (void)ws_size;
    const float* x     = (const float*)d_in[0];
    const int*   pos   = (const int*)  d_in[1];
    const float* freq  = (const float*)d_in[2];
    const float* phase = (const float*)d_in[3];
    const float* aw    = (const float*)d_in[4];
    const float* ab    = (const float*)d_in[5];
    float* out = (float*)d_out;

    dim3 grid(TOKENS / TOK_PER_BLOCK);   // 2048 blocks x 256 threads (8 waves)
    afpe_kernel<<<grid, dim3(256), 0, stream>>>(x, pos, freq, phase, aw, ab, out);
}
